// CnnAttention_16612933500995
// MI455X (gfx1250) — matrix operand order repaired and checked by reading, not compiled, not run on hardware
//
#include <hip/hip_runtime.h>
#include <hip/hip_bf16.h>

// ---------------------------------------------------------------------------
// CDNA5 (gfx1250) fused CNN-attention pipeline, bf16 WMMA + f32 accumulate.
// All WMMA operands are kept in fragment-major layout (per-lane 16 bf16
// contiguous = 32B) so fragment loads are 2x b128 instead of 16 scalar
// gathers. prep pre-swizzles weights; the conv epilogue emits K/V in
// B-fragment order and attention emits newv in A-fragment order.
//
// Fragment conventions (ISA 7.12.2, wave32):
//  A 16x32 bf16 : lane = (row&15) + 16*((k>>3)&1), elem e = (k&7) + 8*(k>>4)
//  B 32x16 bf16 : lane = (col&15) + 16*((k>>3)&1), e = (k&7) + 8*(k>>4); attn K/V: k = 16*(lane>>4)+e
//  C/D 16x16 f32: vgpr r -> row = r + 8*(lane>=16), col = lane&15
// ---------------------------------------------------------------------------

typedef __attribute__((ext_vector_type(16))) __bf16 v16bf;
typedef __attribute__((ext_vector_type(8)))  __bf16 bf16x8;
typedef __attribute__((ext_vector_type(8)))  float  v8f;

#define B_   2
#define L_   2048
#define D_   512
#define H_   8
#define HD_  64
#define O_   1536   // 3*H*HD
#define NT_  4096   // B*L

static __device__ __forceinline__ v8f zero8() {
  v8f z;
#pragma unroll
  for (int i = 0; i < 8; ++i) z[i] = 0.0f;
  return z;
}

static __device__ __forceinline__ v8f wmma_bf16(v16bf a, v16bf b, v8f c) {
  return __builtin_amdgcn_wmma_f32_16x16x32_bf16(false, a, false, b,
                                                 (short)0, c, false, false);
}

static __device__ __forceinline__ float swishf(float v) {
  return v / (1.0f + __expf(-v));
}

// Slow path gather (used only for the once-per-kernel Q fragment load).
static __device__ __forceinline__ v16bf load_a_frag_gather(const __bf16* p,
                                                           int ld) {
  const int lane = threadIdx.x & 31;
  const int row  = lane & 15;
  const int klo  = (lane & 16) ? 16 : 0;
  const __bf16* r = p + (size_t)row * ld;
  v16bf a;
#pragma unroll
  for (int e = 0; e < 16; ++e) a[e] = r[klo + e];
  return a;
}

// ---------------------------------------------------------------------------
// Kernel 0: weight prep into fragment-major bf16.
//  wkB : conv weights, [kchunk(48)][ntile(96)][lane(32)][e(16)]
//        element: k = kc*32 + 8*(lane>>4) + (e&7) + 16*(e>>3) (k = tap*512 + i), n = channel
//  fcwB: fc weights,   [kchunk(16)][ntile(32)][lane(32)][e(16)]
// ---------------------------------------------------------------------------
__global__ __launch_bounds__(256) void prep_kernel(
    const float* __restrict__ cnn_w, const float* __restrict__ fc_w,
    __bf16* __restrict__ wkB, __bf16* __restrict__ fcwB) {
  const int idx = blockIdx.x * 256 + threadIdx.x;
  const int NW = O_ * O_;  // 1536 K * 1536 N = 2,359,296
  if (idx < NW) {
    int e = idx & 15;
    int lane = (idx >> 4) & 31;
    int t2 = idx >> 9;
    int ntile = t2 % 96;
    int kc = t2 / 96;
    int k = kc * 32 + ((lane >> 4) << 3) + (e & 7) + ((e >> 3) << 4);   // 0..1535
    int n = ntile * 16 + (lane & 15);           // 0..1535
    int tap = k >> 9, i = k & 511;
    wkB[idx] = (__bf16)cnn_w[((size_t)n * D_ + i) * 3 + tap];
  } else {
    int j = idx - NW;
    if (j < D_ * D_) {
      int e = j & 15;
      int lane = (j >> 4) & 31;
      int t2 = j >> 9;
      int ntile = t2 & 31;
      int kc = t2 >> 5;
      int k = kc * 32 + ((lane >> 4) << 3) + (e & 7) + ((e >> 3) << 4);
      int o = ntile * 16 + (lane & 15);
      fcwB[j] = (__bf16)fc_w[(size_t)o * D_ + k];
    }
  }
}

// ---------------------------------------------------------------------------
// Kernel 1: conv1d(k=3,pad=1)+bias+swish as 4096x1536x1536 bf16 GEMM.
// Block: 128 tokens x 128 channels, 8 waves (4x2), 32x64 per wave.
// A staged through LDS in A-fragment order (f32->bf16 + edge zero-pad);
// B staged through LDS as a straight 8KB memcpy (already fragment-major).
// Epilogue scatters Q row-major, K/V in B-fragment order for attention.
// ---------------------------------------------------------------------------
__global__ __launch_bounds__(256) void conv_qkv_kernel(
    const float* __restrict__ x, const __bf16* __restrict__ wkB,
    const float* __restrict__ cnn_b,
    __bf16* __restrict__ Q, __bf16* __restrict__ Kf, __bf16* __restrict__ Vf) {
  __shared__ __align__(16) __bf16 Af[8][32][16];  // 8 row-blocks, frag order
  __shared__ __align__(16) __bf16 Bf[8][32][16];  // 8 n-tiles,   frag order
  const int tid = threadIdx.x;
  const int t0 = blockIdx.x * 128;
  const int n0 = blockIdx.y * 128;
  const int wid = tid >> 5, lane = tid & 31;
  const int wm = wid & 3, wn = wid >> 2;

  v8f acc[2][4];
#pragma unroll
  for (int i = 0; i < 2; ++i)
#pragma unroll
    for (int j = 0; j < 4; ++j) acc[i][j] = zero8();

  for (int kt = 0; kt < 3 * D_; kt += 32) {
    const int kc  = kt >> 5;
    const int ksh = kt >> 9;        // conv tap 0..2
    const int i0  = kt & 511;       // channel offset within tap
    {   // stage A: 16 consecutive k for one token row -> two 16B frag stores
      int row  = tid >> 1;
      int part = (tid & 1) << 4;    // k offset 0 or 16
      int t = t0 + row;
      int b = t >> 11, l = t & 2047;
      int lsrc = l + ksh - 1;
      bf16x8 lo, hi;
      if (lsrc >= 0 && lsrc < L_) {
        const float* src = x + ((size_t)(b * L_ + lsrc) * D_ + i0 + part);
#pragma unroll
        for (int e = 0; e < 8; ++e) {
          lo[e] = (__bf16)src[e];
          hi[e] = (__bf16)src[e + 8];
        }
      } else {
#pragma unroll
        for (int e = 0; e < 8; ++e) { lo[e] = (__bf16)0.0f; hi[e] = (__bf16)0.0f; }
      }
      int blk = row >> 4;
      int lane_lo = row & 15;
      int eoff = (part >> 4) << 3;  // 0 or 8
      *(bf16x8*)&Af[blk][lane_lo][eoff]      = lo;  // k part+0..7
      *(bf16x8*)&Af[blk][lane_lo + 16][eoff] = hi;  // k part+8..15
    }
    {   // stage B: contiguous 8KB copy (already fragment-major in global)
      const v16bf* src =
          (const v16bf*)(wkB + (size_t)((kc * 96) + (n0 >> 4)) * 512);
      ((v16bf*)Bf)[tid] = src[tid];
    }
    __syncthreads();
    v16bf af[2];
#pragma unroll
    for (int mt = 0; mt < 2; ++mt)
      af[mt] = *(const v16bf*)&Af[wm * 2 + mt][lane][0];
#pragma unroll
    for (int nt = 0; nt < 4; ++nt) {
      v16bf bf = *(const v16bf*)&Bf[wn * 4 + nt][lane][0];
#pragma unroll
      for (int mt = 0; mt < 2; ++mt)
        acc[mt][nt] = wmma_bf16(af[mt], bf, acc[mt][nt]);
    }
    __syncthreads();
  }

  // Epilogue: bias + swish, scatter into q/k/v (K,V in B-fragment order).
  const int hl   = (lane >> 4) & 1;
  const int ncol = lane & 15;
#pragma unroll
  for (int mt = 0; mt < 2; ++mt)
#pragma unroll
    for (int nt = 0; nt < 4; ++nt) {
#pragma unroll
      for (int r = 0; r < 8; ++r) {
        int m = wm * 32 + mt * 16 + r + hl * 8;
        int n = n0 + wn * 64 + nt * 16 + ncol;
        float v = swishf(acc[mt][nt][r] + cnn_b[n]);
        int t = t0 + m;
        int b = t >> 11, l = t & 2047;
        int h = n / 192, j = n % 192;
        size_t bh = (size_t)(b * H_ + h);
        if (j < 64) {
          Q[(bh * L_ + l) * HD_ + j] = (__bf16)v;
        } else if (j < 128) {
          int d = j - 64;  // Kf[bh][mtile=l>>4][dchunk=d>>5][lane][e]
          size_t fi = (((bh * 128 + (l >> 4)) * 2 + (d >> 5)) * 32 +
                       ((l & 15) + (((d >> 4) & 1) << 4))) * 16 + (d & 15);
          Kf[fi] = (__bf16)v;
        } else {
          int d = j - 128; // Vf[bh][mchunk=l>>5][dtile=d>>4][lane][e]
          size_t fi = (((bh * 64 + (l >> 5)) * 4 + (d >> 4)) * 32 +
                       ((d & 15) + (((l >> 4) & 1) << 4))) * 16 + (l & 15);
          Vf[fi] = (__bf16)v;
        }
      }
    }
}

// ---------------------------------------------------------------------------
// Kernel 2: flash attention, one wave = 16 query rows of one (b,h).
// K/V arrive in B-fragment order (vector loads). Softmax is register-resident
// (xor-shuffle row reductions inside each 16-lane half); only the P bf16
// transpose to A-fragment order touches LDS. Writes newv in fc A-frag order.
// ---------------------------------------------------------------------------
__global__ __launch_bounds__(256) void attn_kernel(
    const __bf16* __restrict__ Q, const __bf16* __restrict__ Kf,
    const __bf16* __restrict__ Vf, __bf16* __restrict__ newvF) {
  __shared__ __align__(16) __bf16 Pf[8][2][32][16];  // per wave: 2 A-frag tiles
  const int tid  = threadIdx.x;
  const int wid  = tid >> 5, lane = tid & 31;
  const int bh   = blockIdx.y;
  const int b    = bh >> 3, h = bh & 7;
  const int l0   = blockIdx.x * 128 + wid * 16;
  const int hl   = (lane >> 4) & 1;
  const int ncol = lane & 15;
  const float scale = 0.02209708691207961f;  // 1/sqrt(L=2048)

  v16bf aq[2];
#pragma unroll
  for (int ks = 0; ks < 2; ++ks)
    aq[ks] = load_a_frag_gather(Q + ((size_t)bh * L_ + l0) * HD_ + ks * 32, HD_);

  v8f acc[4];
#pragma unroll
  for (int nt = 0; nt < 4; ++nt) acc[nt] = zero8();
  float m_i[8], l_i[8];
#pragma unroll
  for (int r = 0; r < 8; ++r) { m_i[r] = -3.0e38f; l_i[r] = 0.0f; }

  for (int m0 = 0; m0 < L_; m0 += 64) {
    // S = Q K (unscaled in registers; scale folded into softmax)
    v8f st[4];
#pragma unroll
    for (int nt = 0; nt < 4; ++nt) {
      st[nt] = zero8();
#pragma unroll
      for (int ks = 0; ks < 2; ++ks) {
        v16bf bk = *(const v16bf*)(Kf +
            ((((size_t)bh * 128 + (m0 >> 4) + nt) * 2 + ks) * 32 + lane) * 16);
        st[nt] = wmma_bf16(aq[ks], bk, st[nt]);
      }
    }
    // Row maxima: per-lane partial over 4 tiles, then xor-shuffle across the
    // 16-lane half that holds the row.
    float rmax[8], corr[8], psum[8];
#pragma unroll
    for (int r = 0; r < 8; ++r) {
      float v = st[0][r];
#pragma unroll
      for (int nt = 1; nt < 4; ++nt) v = fmaxf(v, st[nt][r]);
      rmax[r] = v * scale;
    }
#pragma unroll
    for (int r = 0; r < 8; ++r) {
#pragma unroll
      for (int mask = 1; mask < 16; mask <<= 1)
        rmax[r] = fmaxf(rmax[r], __shfl_xor(rmax[r], mask, 32));
    }
#pragma unroll
    for (int r = 0; r < 8; ++r) {
      float mnew = fmaxf(m_i[r], rmax[r]);
      corr[r] = __expf(m_i[r] - mnew);
      m_i[r] = mnew;
      psum[r] = 0.0f;
    }
    // P = exp(S*scale - m), streamed to LDS in A-fragment order.
#pragma unroll
    for (int nt = 0; nt < 4; ++nt) {
      int k = nt * 16 + ncol;       // local key index 0..63
      int ks = k >> 5, k32 = k & 31;
      int lp_hi = (k32 >> 4) << 4;
      int e = k32 & 15;
#pragma unroll
      for (int r = 0; r < 8; ++r) {
        float p = __expf(st[nt][r] * scale - m_i[r]);
        psum[r] += p;
        Pf[wid][ks][(r + 8 * hl) + lp_hi][e] = (__bf16)p;
      }
    }
#pragma unroll
    for (int r = 0; r < 8; ++r) {
#pragma unroll
      for (int mask = 1; mask < 16; mask <<= 1)
        psum[r] += __shfl_xor(psum[r], mask, 32);
      l_i[r] = l_i[r] * corr[r] + psum[r];
    }
#pragma unroll
    for (int nt = 0; nt < 4; ++nt)
#pragma unroll
      for (int r = 0; r < 8; ++r) acc[nt][r] *= corr[r];
    __syncthreads();
    // O += P V
    v16bf ap[2];
    ap[0] = *(const v16bf*)&Pf[wid][0][lane][0];
    ap[1] = *(const v16bf*)&Pf[wid][1][lane][0];
#pragma unroll
    for (int nt = 0; nt < 4; ++nt) {
#pragma unroll
      for (int ks = 0; ks < 2; ++ks) {
        v16bf bv = *(const v16bf*)(Vf +
            ((((size_t)bh * 64 + (m0 >> 5) + ks) * 4 + nt) * 32 + lane) * 16);
        acc[nt] = wmma_bf16(ap[ks], bv, acc[nt]);
      }
    }
    __syncthreads();
  }

  // Normalize and emit newv in the fc kernel's A-fragment order.
#pragma unroll
  for (int nt = 0; nt < 4; ++nt) {
#pragma unroll
    for (int r = 0; r < 8; ++r) {
      float o = acc[nt][r] / l_i[r];
      int t = l0 + r + 8 * hl;
      int ch = h * HD_ + nt * 16 + ncol;
      size_t fi = (((size_t)(t >> 4) * 16 + (ch >> 5)) * 32 +
                   ((t & 15) + (((ch >> 3) & 1) << 4))) * 16 +
                  ((ch & 7) + (((ch >> 4) & 1) << 3));
      newvF[fi] = (__bf16)o;
    }
  }
}

// ---------------------------------------------------------------------------
// Kernel 3: m = swish(newv @ fc_w^T + fc_b), 4096x512x512 bf16 GEMM.
// Both operands already fragment-major in global memory -> no LDS, no
// barriers; pure vector loads + WMMA.
// ---------------------------------------------------------------------------
__global__ __launch_bounds__(256) void fc_kernel(
    const __bf16* __restrict__ newvF, const __bf16* __restrict__ fcwB,
    const float* __restrict__ fc_b, float* __restrict__ mout) {
  const int tid = threadIdx.x;
  const int t0 = blockIdx.x * 128;
  const int n0 = blockIdx.y * 128;
  const int wid = tid >> 5, lane = tid & 31;
  const int wm = wid & 3, wn = wid >> 2;

  v8f acc[2][4];
#pragma unroll
  for (int i = 0; i < 2; ++i)
#pragma unroll
    for (int j = 0; j < 4; ++j) acc[i][j] = zero8();

  for (int kc = 0; kc < 16; ++kc) {   // K = 512 in chunks of 32
    v16bf af[2];
#pragma unroll
    for (int mt = 0; mt < 2; ++mt)
      af[mt] = *(const v16bf*)(newvF +
          (((size_t)((t0 >> 4) + wm * 2 + mt) * 16 + kc) * 32 + lane) * 16);
#pragma unroll
    for (int nt = 0; nt < 4; ++nt) {
      v16bf bf = *(const v16bf*)(fcwB +
          (((size_t)kc * 32 + (n0 >> 4) + wn * 4 + nt) * 32 + lane) * 16);
#pragma unroll
      for (int mt = 0; mt < 2; ++mt)
        acc[mt][nt] = wmma_bf16(af[mt], bf, acc[mt][nt]);
    }
  }

  const int hl   = (lane >> 4) & 1;
  const int ncol = lane & 15;
#pragma unroll
  for (int mt = 0; mt < 2; ++mt)
#pragma unroll
    for (int nt = 0; nt < 4; ++nt) {
#pragma unroll
      for (int r = 0; r < 8; ++r) {
        int m = wm * 32 + mt * 16 + r + hl * 8;
        int n = n0 + wn * 64 + nt * 16 + ncol;
        float v = acc[mt][nt][r] + fc_b[n];
        mout[(size_t)(t0 + m) * 512 + n] = swishf(v);
      }
    }
}

// ---------------------------------------------------------------------------
// Kernel 4: out = layer_norm(2*x + m). One 256-thread block per token row.
// ---------------------------------------------------------------------------
__global__ __launch_bounds__(256) void ln_kernel(
    const float* __restrict__ x, const float* __restrict__ mbuf,
    float* __restrict__ out) {
  __shared__ float rs[256], rq[256];
  const int t = blockIdx.x;
  const int tid = threadIdx.x;
  const size_t base = (size_t)t * 512;
  float a0 = 2.0f * x[base + tid] + mbuf[base + tid];
  float a1 = 2.0f * x[base + tid + 256] + mbuf[base + tid + 256];
  rs[tid] = a0 + a1;
  rq[tid] = a0 * a0 + a1 * a1;
  __syncthreads();
  for (int off = 128; off > 0; off >>= 1) {
    if (tid < off) { rs[tid] += rs[tid + off]; rq[tid] += rq[tid + off]; }
    __syncthreads();
  }
  float mu  = rs[0] * (1.0f / 512.0f);
  float var = rq[0] * (1.0f / 512.0f) - mu * mu;
  float inv = rsqrtf(var + 1e-5f);
  out[base + tid]       = (a0 - mu) * inv;
  out[base + tid + 256] = (a1 - mu) * inv;
}

// ---------------------------------------------------------------------------
// Workspace: wkB 4.5MB | fcwB 0.5MB | Q 4MB | Kf 4MB | Vf 4MB | newvF 4MB |
// m 8MB  => ~29MB.
// ---------------------------------------------------------------------------
extern "C" void kernel_launch(void* const* d_in, const int* in_sizes, int n_in,
                              void* d_out, int out_size, void* d_ws,
                              size_t ws_size, hipStream_t stream) {
  (void)in_sizes; (void)n_in; (void)out_size; (void)ws_size;
  const float* x     = (const float*)d_in[0];
  const float* cnn_w = (const float*)d_in[1];
  const float* cnn_b = (const float*)d_in[2];
  const float* fc_w  = (const float*)d_in[3];
  const float* fc_b  = (const float*)d_in[4];
  float* out = (float*)d_out;
  char*  ws  = (char*)d_ws;

  __bf16* wkB   = (__bf16*)(ws);
  __bf16* fcwB  = (__bf16*)(ws + 4718592);
  __bf16* Q     = (__bf16*)(ws + 5242880);
  __bf16* Kf    = (__bf16*)(ws + 9437184);
  __bf16* Vf    = (__bf16*)(ws + 13631488);
  __bf16* newvF = (__bf16*)(ws + 17825792);
  float*  mbuf  = (float*)(ws + 22020096);

  prep_kernel<<<dim3(10240), dim3(256), 0, stream>>>(cnn_w, fc_w, wkB, fcwB);
  conv_qkv_kernel<<<dim3(32, 12), dim3(256), 0, stream>>>(x, wkB, cnn_b, Q, Kf, Vf);
  attn_kernel<<<dim3(16, 16), dim3(256), 0, stream>>>(Q, Kf, Vf, newvF);
  fc_kernel<<<dim3(32, 4), dim3(256), 0, stream>>>(newvF, fcwB, fc_b, mbuf);
  ln_kernel<<<dim3(4096), dim3(256), 0, stream>>>(x, mbuf, out);
}
